// EncoderTact_61924838474026
// MI455X (gfx1250) — matrix operand order repaired and checked by reading, compiled, not run on hardware
//
#include <hip/hip_runtime.h>

// SNN alpha-PSP + linear + refractory spike scan for MI455X (gfx1250, wave32).
//
// Reference: psp = eps (*) x ; u = W*psp ; sequential spike scan.
// eps-filter and W commute (both linear): GEMM first (156ch -> 50ch), then the
// alpha filter as an exact 2nd-order IIR recurrence (+2 truncation-correction
// taps) fused into the refractory scan.
//
// Stage 1: y[b*T+t, o] = sum_c x[b,c,t] * W[o,c]
//   - W staged in LDS as f16 [64][160] once per block (ds_load_b128 fragments)
//   - one wave per 16-row M tile, 4 accumulators cover all 50 out channels
//   - v_wmma_f32_16x16x32_f16, K tail handled by clamp+select (no EXEC branches)
// Stage 2: per (b,o) lane: IIR filter + 32-bit spike-history refractory scan,
//   time loop split into branch-free segments.

typedef __attribute__((ext_vector_type(16))) _Float16 v16h;
typedef __attribute__((ext_vector_type(8)))  float    v8f;

#define B_    512
#define CIN   156
#define COUT  50
#define T_    325
#define KREF  32
#define THETA 10.0f

#define BT      (B_ * T_)        // 166400 rows = exactly 10400 16-row tiles
#define MTILES  (BT / 16)        // 10400
#define NPAD    64               // 50 -> 4 N tiles
#define CPAD    160              // 156 -> 5 K chunks of 32
#define KCHUNKS 5
#define WVB     8                // waves per 256-thread block (wave32)

__global__ __launch_bounds__(256)
void snn_gemm_wmma(const float* __restrict__ x,   // [B, CIN, T]
                   const float* __restrict__ w,   // [COUT, CIN]
                   float* __restrict__ y)         // [B*T, COUT]
{
    __shared__ __align__(32) _Float16 wlds[NPAD * CPAD];   // 20 KB

    // Cooperative W stage: f32 -> f16, zero-padded. Clamped addresses +
    // value select -> no divergent exec around loads.
    for (int i = threadIdx.x; i < NPAD * CPAD; i += 256) {
        const int o  = i / CPAD;
        const int c  = i - o * CPAD;
        const int oc = (o < COUT) ? o : (COUT - 1);
        const int cc = (c < CIN)  ? c : (CIN - 1);
        float v = w[oc * CIN + cc];
        v = (o < COUT && c < CIN) ? v : 0.0f;
        wlds[i] = (_Float16)v;
    }
    __syncthreads();

    const int lane = threadIdx.x & 31;
    const int wid  = threadIdx.x >> 5;
    const int m_tile = blockIdx.x * WVB + wid;     // wave-uniform
    if (m_tile >= MTILES) return;                  // uniform -> EXEC stays full

    const int half = lane >> 4;                    // K half-lane split
    const int l16  = lane & 15;

    // A-matrix lane row: r = b*T + t
    const int r = m_tile * 16 + l16;
    const int b = r / T_;
    const int t = r - b * T_;
    const float* xrow = x + (size_t)b * CIN * T_ + t;   // element c at c*T_

    v8f acc[4] = {v8f{}, v8f{}, v8f{}, v8f{}};

    #pragma unroll
    for (int kc = 0; kc < KCHUNKS; ++kc) {
        const int k0 = kc * 32;

        // A 16x32 f16 fragment: elem i -> K = 16*half + i.
        // Tail chunk: clamp address, zero via select (branch-free).
        v16h afrag;
        #pragma unroll
        for (int i = 0; i < 16; ++i) {
            const int K  = half * 16 + i;
            const int c  = k0 + K;
            const int cc = (c < CIN) ? c : (CIN - 1);
            float v = xrow[(size_t)cc * T_];
            v = (c < CIN) ? v : 0.0f;
            afrag[i] = (_Float16)v;
        }

        // B 32x16 f16 fragments from LDS: elem i -> K = 16*half + i, so each
        // fragment is 16 contiguous f16 (32 B, aligned) -> 2x ds_load_b128.
        const int cb = k0 + half * 16;
        v16h bfrag[4];
        #pragma unroll
        for (int n = 0; n < 4; ++n)
            bfrag[n] = *(const v16h*)&wlds[(n * 16 + l16) * CPAD + cb];

        #pragma unroll
        for (int n = 0; n < 4; ++n)
            acc[n] = __builtin_amdgcn_wmma_f32_16x16x32_f16(
                         false, afrag, false, bfrag[n], (short)0, acc[n],
                         false, false);
    }

    // C/D f32 layout: lane col N = lane%16, VGPR v -> row M = v + 8*half
    const int rbase = m_tile * 16;
    #pragma unroll
    for (int n = 0; n < 4; ++n) {
        const int o = n * 16 + l16;
        if (o < COUT) {
            #pragma unroll
            for (int v = 0; v < 8; ++v)
                y[(size_t)(rbase + v + 8 * half) * COUT + o] = acc[n][v];
        }
    }
}

// Exact alpha-filter IIR (tau_sr = 10, a = exp(-0.1), c = e/10):
//   u[t] = 2a u[t-1] - a^2 u[t-2] + c*a*y[t-1]
//          - 100*c*a^100 * y[t-100] + 99*c*a^101 * y[t-101]
__global__ __launch_bounds__(256)
void snn_scan(const float* __restrict__ y,    // [B*T, COUT]
              float* __restrict__ out)        // [B, COUT, T]
{
    const int p = blockIdx.x * blockDim.x + threadIdx.x;
    if (p >= B_ * COUT) return;
    const int b = p / COUT;
    const int o = p - b * COUT;

    // nu[j] = -SCALE_REF*THETA * (j/tau_ref) * exp(1 - j) ; tau_ref = 1
    float nu[KREF];
    #pragma unroll
    for (int j = 1; j < KREF; ++j)
        nu[j] = -20.0f * (float)j * __expf(1.0f - (float)j);

    const float A1   = 1.8096748360719190f;    // 2*exp(-1/10)
    const float A2   = 0.8187307530779818f;    // exp(-2/10)
    const float CA   = 0.2459603111156950f;    // (e/10)*exp(-1/10)
    const float C100 = 1.2340980408667956e-3f; // 100*(e/10)*a^100
    const float C101 = 1.1054907968112316e-3f; //  99*(e/10)*a^101

    const float* yp = y + (size_t)b * T_ * COUT + o;   // row stride COUT
    float* op = out + ((size_t)b * COUT + o) * T_;

    float u1 = 0.0f, u2 = 0.0f;
    unsigned hist = 0u;                         // bit (j-1) = spike at t-j

    auto step = [&](float u) -> float {
        float refr = 0.0f;
        #pragma unroll
        for (int j = 1; j < KREF; ++j)
            refr += ((hist >> (j - 1)) & 1u) ? nu[j] : 0.0f;
        const bool sp = (u + refr >= THETA);
        hist = (hist << 1) | (sp ? 1u : 0u);
        return sp ? 1.0f : 0.0f;
    };

    // t = 0: u = 0 (eps[0] = 0), no history -> never spikes
    op[0] = 0.0f;                               // hist stays 0

    for (int t = 1; t < 100; ++t) {             // no correction taps yet
        const float u = A1 * u1 - A2 * u2 + CA * yp[(size_t)(t - 1) * COUT];
        u2 = u1; u1 = u;
        op[t] = step(u);
    }
    {                                           // t = 100: first tap enters
        const float u = A1 * u1 - A2 * u2 + CA * yp[(size_t)99 * COUT]
                      - C100 * yp[0];
        u2 = u1; u1 = u;
        op[100] = step(u);
    }
    for (int t = 101; t < T_; ++t) {            // steady state, branch-free
        const float u = A1 * u1 - A2 * u2
                      + CA   * yp[(size_t)(t - 1)   * COUT]
                      - C100 * yp[(size_t)(t - 100) * COUT]
                      + C101 * yp[(size_t)(t - 101) * COUT];
        u2 = u1; u1 = u;
        op[t] = step(u);
    }
}

extern "C" void kernel_launch(void* const* d_in, const int* in_sizes, int n_in,
                              void* d_out, int out_size, void* d_ws, size_t ws_size,
                              hipStream_t stream) {
    const float* x = (const float*)d_in[0];   // [512,156,325] f32
    const float* w = (const float*)d_in[1];   // [50,156] f32
    float* out = (float*)d_out;               // [512,50,325] f32
    float* y   = (float*)d_ws;                // B*T*COUT*4 = 33.28 MB scratch

    dim3 blk(256);
    snn_gemm_wmma<<<dim3(MTILES / WVB), blk, 0, stream>>>(x, w, y);
    snn_scan<<<dim3((B_ * COUT + 255) / 256), blk, 0, stream>>>(y, out);
}
